// ICON_UNCROPPED_53850299958082
// MI455X (gfx1250) — matrix operand order repaired and checked by reading, not compiled, not run on hardware
//
#include <hip/hip_runtime.h>
#include <math.h>

// ---------------- model constants ----------------
#define BS    8
#define PAIRS 5
#define CCH   2
#define PNUM  8
#define RES   16
#define HW    128          // PNUM*RES
#define DTOK  512
#define NHEAD 8
#define DH    64           // DTOK/NHEAD
#define DFF   2048
#define NLAY  6
#define SEQ   640          // PAIRS*2*PNUM*PNUM
#define NTOK  (BS*SEQ)     // 5120 token rows
#define PATCH 512          // CCH*RES*RES

typedef _Float16 v16h __attribute__((ext_vector_type(16)));
typedef _Float16 v8h  __attribute__((ext_vector_type(8)));
typedef float    v8f  __attribute__((ext_vector_type(8)));

// ---------------- WMMA helpers ----------------
__device__ inline v8f wmma_f16(v16h a, v16h b, v8f c) {
  return __builtin_amdgcn_wmma_f32_16x16x32_f16(
      /*neg_a=*/false, a, /*neg_b=*/false, b,
      /*c_mod=*/(short)0, c, /*reuse_a=*/false, /*reuse_b=*/false);
}

// A fragment (16x32): lane m=lane&15; lanes<16: K {kb..kb+7, kb+16..kb+23},
// lanes>=16: K {kb+8..kb+15, kb+24..kb+31}
__device__ inline v16h load_a(const _Float16* rowptr, int kb, int koff) {
  v8h lo = *(const v8h*)(rowptr + kb + koff);
  v8h hi = *(const v8h*)(rowptr + kb + 16 + koff);
  v16h r;
#pragma unroll
  for (int i = 0; i < 8; ++i) { r[i] = lo[i]; r[i + 8] = hi[i]; }
  return r;
}

// B fragment (32x16, fed as Bt[N][K] row-major): lane n=lane&15;
// lanes<16: K {kb..kb+7, kb+16..kb+23}; lanes>=16: K {kb+8..kb+15, kb+24..kb+31}
__device__ inline v16h load_b(const _Float16* rowptr, int kb, int lane) {
  return load_a(rowptr, kb, (lane >> 4) << 3);
}

// one wave computes a 32x64 f32 tile (2x4 grid of 16x16 WMMA tiles) of
// A[M,K] * W[N,K]^T.  All 6 fragment loads are hoisted ahead of the 8 WMMAs
// per K-step so they clause together and overlap with matrix issue.
__device__ inline void gemm_tile_2x4(const _Float16* __restrict__ A,
                                     const _Float16* __restrict__ W,
                                     int K, int tm, int tn, int lane,
                                     v8f acc[2][4]) {
  const int mr   = lane & 15;
  const int koff = (lane >> 4) << 3;
  const _Float16* ar0 = A + (size_t)(tm * 32 + mr) * K;
  const _Float16* ar1 = ar0 + (size_t)16 * K;
  const _Float16* br  = W + (size_t)(tn * 64 + mr) * K;
  for (int kb = 0; kb < K; kb += 32) {
    if (kb + 128 < K) {                       // global_prefetch_b8 path
      __builtin_prefetch(ar0 + kb + 128, 0, 2);
      __builtin_prefetch(ar1 + kb + 128, 0, 2);
      __builtin_prefetch(br  + kb + 128, 0, 2);
    }
    v16h a0 = load_a(ar0, kb, koff);
    v16h a1 = load_a(ar1, kb, koff);
    v16h b0 = load_b(br,                      kb, lane);
    v16h b1 = load_b(br + (size_t)16 * K,     kb, lane);
    v16h b2 = load_b(br + (size_t)32 * K,     kb, lane);
    v16h b3 = load_b(br + (size_t)48 * K,     kb, lane);
    acc[0][0] = wmma_f16(a0, b0, acc[0][0]);
    acc[1][0] = wmma_f16(a1, b0, acc[1][0]);
    acc[0][1] = wmma_f16(a0, b1, acc[0][1]);
    acc[1][1] = wmma_f16(a1, b1, acc[1][1]);
    acc[0][2] = wmma_f16(a0, b2, acc[0][2]);
    acc[1][2] = wmma_f16(a1, b2, acc[1][2]);
    acc[0][3] = wmma_f16(a0, b3, acc[0][3]);
    acc[1][3] = wmma_f16(a1, b3, acc[1][3]);
  }
}

// ---------------- elementwise kernels ----------------
__global__ void f32_to_f16_kernel(const float* __restrict__ in,
                                  _Float16* __restrict__ out, int n) {
  int i = blockIdx.x * blockDim.x + threadIdx.x;
  if (i < n) out[i] = (_Float16)in[i];
}

__global__ void patchify_kernel(const float* __restrict__ initp,
                                const float* __restrict__ endp,
                                _Float16* __restrict__ A0, int total) {
  int i = blockIdx.x * blockDim.x + threadIdx.x;
  if (i >= total) return;
  int j = i & 511, row = i >> 9;
  int b = row / SEQ, sr = row - b * SEQ;
  int fi = sr >> 6, p = sr & 63;
  int pair = fi >> 1, which = fi & 1;
  int c = j >> 8, r1 = (j >> 4) & 15, r2 = j & 15;
  int y = ((p >> 3) << 4) + r1, x = ((p & 7) << 4) + r2;
  const float* src = which ? endp : initp;
  float v = src[(((size_t)(b * PAIRS + pair) * CCH + c) * HW + y) * HW + x];
  A0[i] = (_Float16)v;
}

__global__ void add_pos_kernel(const float* __restrict__ tmp,
                               const float* __restrict__ ppos,
                               const float* __restrict__ fpos,
                               float* __restrict__ F,
                               _Float16* __restrict__ F16, int total) {
  int i = blockIdx.x * blockDim.x + threadIdx.x;
  if (i >= total) return;
  int d = i & 511, row = i >> 9;
  int sr = row % SEQ, p = sr & 63, fi = sr >> 6;
  float v = tmp[i] + ppos[p * DTOK + d] + fpos[fi * DTOK + d];
  F[i] = v; F16[i] = (_Float16)v;
}

__global__ void pack_init_kernel(const _Float16* __restrict__ F16,
                                 _Float16* __restrict__ Apost, int total) {
  int i = blockIdx.x * blockDim.x + threadIdx.x;
  if (i >= total) return;
  int d = i & 511, row = i >> 9;             // row = (b*PAIRS+pair)*64+p
  int p = row & 63, bp = row >> 6;
  int pair = bp % PAIRS, b = bp / PAIRS;
  int srow = b * SEQ + (pair * 2) * 64 + p;  // init slot
  Apost[i] = F16[(size_t)srow * DTOK + d];
}

__global__ void depatch_kernel(const float* __restrict__ T,
                               float* __restrict__ out, int total) {
  int i = blockIdx.x * blockDim.x + threadIdx.x;
  if (i >= total) return;
  int x = i & 127, y = (i >> 7) & 127, c = (i >> 14) & 1, bp = i >> 15;
  int p = ((y >> 4) << 3) + (x >> 4);
  int j = (c << 8) + ((y & 15) << 4) + (x & 15);
  out[i] = T[((size_t)bp * 64 + p) * DTOK + j];
}

// f = LN(f + add) * s + b ; wave32 per row, 16 elems/lane
__global__ void ln_res_kernel(float* __restrict__ F,
                              const float* __restrict__ add,
                              const float* __restrict__ s,
                              const float* __restrict__ bsh,
                              _Float16* __restrict__ F16) {
  int row = blockIdx.x, lane = threadIdx.x;
  const float* a = F   + (size_t)row * DTOK;
  const float* t = add + (size_t)row * DTOK;
  float x[16], sum = 0.f;
#pragma unroll
  for (int i = 0; i < 16; ++i) { x[i] = a[lane * 16 + i] + t[lane * 16 + i]; sum += x[i]; }
  for (int o = 16; o > 0; o >>= 1) sum += __shfl_xor(sum, o, 32);
  float mu = sum * (1.0f / DTOK), vs = 0.f;
#pragma unroll
  for (int i = 0; i < 16; ++i) { float dv = x[i] - mu; vs += dv * dv; }
  for (int o = 16; o > 0; o >>= 1) vs += __shfl_xor(vs, o, 32);
  float rinv = rsqrtf(vs * (1.0f / DTOK) + 1e-5f);
#pragma unroll
  for (int i = 0; i < 16; ++i) {
    int d = lane * 16 + i;
    float y = (x[i] - mu) * rinv * s[d] + bsh[d];
    F[(size_t)row * DTOK + d] = y;
    F16[(size_t)row * DTOK + d] = (_Float16)y;
  }
}

// ---------------- GEMM kernels ----------------
// C_f32[M,N] = A_f16[M,K] @ W_f16[N,K]^T + bias ; grid (N/64, M/32), 32 thr
__global__ void gemm_bias_f32(const _Float16* __restrict__ A,
                              const _Float16* __restrict__ W,
                              const float* __restrict__ bias,
                              float* __restrict__ C, int N, int K) {
  int lane = threadIdx.x;
  v8f acc[2][4] = {};
  gemm_tile_2x4(A, W, K, blockIdx.y, blockIdx.x, lane, acc);
  int mbase = (lane >> 4) << 3;
  int nlo = lane & 15;
#pragma unroll
  for (int i = 0; i < 2; ++i)
#pragma unroll
    for (int j = 0; j < 4; ++j) {
      int n = blockIdx.x * 64 + j * 16 + nlo;
      float bv = bias[n];
      int m0 = blockIdx.y * 32 + i * 16 + mbase;
#pragma unroll
      for (int r = 0; r < 8; ++r)
        C[(size_t)(m0 + r) * N + n] = acc[i][j][r] + bv;
    }
}

// f16 output with exact GELU epilogue
__global__ void gemm_bias_gelu_f16(const _Float16* __restrict__ A,
                                   const _Float16* __restrict__ W,
                                   const float* __restrict__ bias,
                                   _Float16* __restrict__ C, int N, int K) {
  int lane = threadIdx.x;
  v8f acc[2][4] = {};
  gemm_tile_2x4(A, W, K, blockIdx.y, blockIdx.x, lane, acc);
  int mbase = (lane >> 4) << 3;
  int nlo = lane & 15;
#pragma unroll
  for (int i = 0; i < 2; ++i)
#pragma unroll
    for (int j = 0; j < 4; ++j) {
      int n = blockIdx.x * 64 + j * 16 + nlo;
      float bv = bias[n];
      int m0 = blockIdx.y * 32 + i * 16 + mbase;
#pragma unroll
      for (int r = 0; r < 8; ++r) {
        float v = acc[i][j][r] + bv;
        float g = 0.5f * v * (1.0f + erff(v * 0.70710678118f));
        C[(size_t)(m0 + r) * N + n] = (_Float16)g;
      }
    }
}

// QKV GEMM with head scatter: Q,K as (b,h,S,dh), V transposed as (b,h,dh,S)
__global__ void gemm_qkv(const _Float16* __restrict__ A,
                         const _Float16* __restrict__ W,
                         const float* __restrict__ bias,
                         _Float16* __restrict__ Qo,
                         _Float16* __restrict__ Ko,
                         _Float16* __restrict__ Vt) {
  int lane = threadIdx.x;
  v8f acc[2][4] = {};
  gemm_tile_2x4(A, W, DTOK, blockIdx.y, blockIdx.x, lane, acc);
  int mbase = (lane >> 4) << 3;
  int nlo = lane & 15;
#pragma unroll
  for (int i = 0; i < 2; ++i)
#pragma unroll
    for (int j = 0; j < 4; ++j) {
      int n = blockIdx.x * 64 + j * 16 + nlo;
      float bv = bias[n];
      int m0 = blockIdx.y * 32 + i * 16 + mbase;
#pragma unroll
      for (int r = 0; r < 8; ++r) {
        int row = m0 + r, b = row / SEQ, sidx = row - b * SEQ;
        float v = acc[i][j][r] + bv;
        if (n < DTOK) {
          int h = n >> 6, d = n & 63;
          Qo[(((size_t)b * NHEAD + h) * SEQ + sidx) * DH + d] = (_Float16)v;
        } else if (n < 2 * DTOK) {
          int nn = n - DTOK, h = nn >> 6, d = nn & 63;
          Ko[(((size_t)b * NHEAD + h) * SEQ + sidx) * DH + d] = (_Float16)v;
        } else {
          int nn = n - 2 * DTOK, h = nn >> 6, d = nn & 63;
          Vt[(((size_t)b * NHEAD + h) * DH + d) * SEQ + sidx] = (_Float16)v;
        }
      }
    }
}

// ---------------- fused attention ----------------
// one wave per (q-tile of 16 rows, head, batch). Block mask handled by
// truncating the k range at ((q0>>6)+1)<<6 (64-token blocks align with tiles).
__global__ void attn_kernel(const _Float16* __restrict__ Q,
                            const _Float16* __restrict__ Kh,
                            const _Float16* __restrict__ Vt,
                            _Float16* __restrict__ O) {
  __shared__ float sc[16 * SEQ];
  const int lane = threadIdx.x;
  const int qt = blockIdx.x, h = blockIdx.y, b = blockIdx.z;
  const int q0 = qt * 16;
  const int alen = ((q0 >> 6) + 1) << 6;       // allowed key length
  const int mr = lane & 15, koff = (lane >> 4) << 3;
  const size_t bh = (size_t)b * NHEAD + h;

  const _Float16* qrow = Q + (bh * SEQ + q0 + mr) * DH;
  v16h qa0 = load_a(qrow, 0, koff);
  v16h qa1 = load_a(qrow, 32, koff);

  // scores = Q K^T * 1/sqrt(dh)
  for (int kt = 0; kt * 16 < alen; ++kt) {
    const _Float16* krow = Kh + (bh * SEQ + kt * 16 + mr) * DH;
    v16h kb0 = load_b(krow, 0, lane);
    v16h kb1 = load_b(krow, 32, lane);
    v8f acc = {};
    acc = wmma_f16(qa0, kb0, acc);
    acc = wmma_f16(qa1, kb1, acc);
    int mbase = (lane >> 4) << 3;
    int ncol = kt * 16 + (lane & 15);
#pragma unroll
    for (int r = 0; r < 8; ++r)
      sc[(mbase + r) * SEQ + ncol] = acc[r] * 0.125f;
  }
  __syncthreads();

  // row softmax: 2 lanes per row
  {
    int row = lane & 15, part = lane >> 4;
    float mx = -3.4e38f;
    for (int c = part; c < alen; c += 2) mx = fmaxf(mx, sc[row * SEQ + c]);
    mx = fmaxf(mx, __shfl_xor(mx, 16, 32));
    float sum = 0.f;
    for (int c = part; c < alen; c += 2) {
      float e = __expf(sc[row * SEQ + c] - mx);
      sc[row * SEQ + c] = e; sum += e;
    }
    sum += __shfl_xor(sum, 16, 32);
    float inv = 1.0f / sum;
    for (int c = part; c < alen; c += 2) sc[row * SEQ + c] *= inv;
  }
  __syncthreads();

  // O = P V  (P from LDS, converted f32->f16; Vt gives contiguous B rows)
  v8f o[4] = {};
  for (int kb = 0; kb < alen; kb += 32) {
    v16h pa;
    const float* s0 = &sc[mr * SEQ + kb + koff];
    const float* s1 = &sc[mr * SEQ + kb + 16 + koff];
#pragma unroll
    for (int i = 0; i < 8; ++i) { pa[i] = (_Float16)s0[i]; pa[i + 8] = (_Float16)s1[i]; }
#pragma unroll
    for (int nt = 0; nt < 4; ++nt) {
      const _Float16* vrow = Vt + (bh * DH + nt * 16 + (lane & 15)) * SEQ;
      v16h vb = load_b(vrow, kb, lane);
      o[nt] = wmma_f16(pa, vb, o[nt]);
    }
  }
  int mbase = (lane >> 4) << 3;
  int ncol = lane & 15;
#pragma unroll
  for (int nt = 0; nt < 4; ++nt)
#pragma unroll
    for (int r = 0; r < 8; ++r)
      O[((size_t)b * SEQ + q0 + mbase + r) * DTOK + h * DH + nt * 16 + ncol] =
          (_Float16)o[nt][r];
}

// ---------------- host launch ----------------
extern "C" void kernel_launch(void* const* d_in, const int* in_sizes, int n_in,
                              void* d_out, int out_size, void* d_ws, size_t ws_size,
                              hipStream_t stream) {
  (void)in_sizes; (void)n_in; (void)out_size; (void)ws_size;
  const float* initp  = (const float*)d_in[0];
  const float* endp   = (const float*)d_in[1];
  /* d_in[2] c_mask unused by reference */
  const float* pre_W  = (const float*)d_in[3];
  const float* pre_b  = (const float*)d_in[4];
  const float* post_W = (const float*)d_in[5];
  const float* post_b = (const float*)d_in[6];
  const float* ppos   = (const float*)d_in[7];
  const float* fpos   = (const float*)d_in[8];
  const float* Wqkv   = (const float*)d_in[9];
  const float* bqkv   = (const float*)d_in[10];
  const float* Wo     = (const float*)d_in[11];
  const float* bo     = (const float*)d_in[12];
  const float* ln1s   = (const float*)d_in[13];
  const float* ln1b   = (const float*)d_in[14];
  const float* ln2s   = (const float*)d_in[15];
  const float* ln2b   = (const float*)d_in[16];
  const float* W1     = (const float*)d_in[17];
  const float* b1     = (const float*)d_in[18];
  const float* W2     = (const float*)d_in[19];
  const float* b2     = (const float*)d_in[20];
  float* out = (float*)d_out;

  // bump allocator over d_ws
  size_t off = 0;
  auto alloc = [&](size_t bytes) -> char* {
    char* p = (char*)d_ws + off;
    off = (off + bytes + 255) & ~(size_t)255;
    return p;
  };
  _Float16* preW16  = (_Float16*)alloc((size_t)DTOK * PATCH * 2);
  _Float16* postW16 = (_Float16*)alloc((size_t)PATCH * DTOK * 2);
  _Float16* Wqkv16  = (_Float16*)alloc((size_t)NLAY * 3 * DTOK * DTOK * 2);
  _Float16* Wo16    = (_Float16*)alloc((size_t)NLAY * DTOK * DTOK * 2);
  _Float16* W116    = (_Float16*)alloc((size_t)NLAY * DFF * DTOK * 2);
  _Float16* W216    = (_Float16*)alloc((size_t)NLAY * DTOK * DFF * 2);
  _Float16* A0      = (_Float16*)alloc((size_t)NTOK * PATCH * 2);
  float*    F       = (float*)   alloc((size_t)NTOK * DTOK * 4);
  _Float16* F16     = (_Float16*)alloc((size_t)NTOK * DTOK * 2);
  float*    TMP     = (float*)   alloc((size_t)NTOK * DTOK * 4);
  _Float16* H116    = (_Float16*)alloc((size_t)NTOK * DFF * 2);
  _Float16* Q16     = (_Float16*)alloc((size_t)BS * NHEAD * SEQ * DH * 2);
  _Float16* K16     = (_Float16*)alloc((size_t)BS * NHEAD * SEQ * DH * 2);
  _Float16* VT16    = (_Float16*)alloc((size_t)BS * NHEAD * DH * SEQ * 2);
  _Float16* O16     = (_Float16*)alloc((size_t)NTOK * DTOK * 2);
  _Float16* APOST   = (_Float16*)alloc((size_t)BS * PAIRS * 64 * DTOK * 2);

  auto cvt = [&](const float* src, _Float16* dst, int n) {
    f32_to_f16_kernel<<<(n + 255) / 256, 256, 0, stream>>>(src, dst, n);
  };
  cvt(pre_W,  preW16,  DTOK * PATCH);
  cvt(post_W, postW16, PATCH * DTOK);
  cvt(Wqkv,   Wqkv16,  NLAY * 3 * DTOK * DTOK);
  cvt(Wo,     Wo16,    NLAY * DTOK * DTOK);
  cvt(W1,     W116,    NLAY * DFF * DTOK);
  cvt(W2,     W216,    NLAY * DTOK * DFF);

  {
    int total = NTOK * PATCH;
    patchify_kernel<<<(total + 255) / 256, 256, 0, stream>>>(initp, endp, A0, total);
  }
  // embed: TMP = A0 @ pre_W^T + pre_b  (M=5120, N=512, K=512)
  gemm_bias_f32<<<dim3(DTOK / 64, NTOK / 32), 32, 0, stream>>>(A0, preW16, pre_b, TMP, DTOK, PATCH);
  {
    int total = NTOK * DTOK;
    add_pos_kernel<<<(total + 255) / 256, 256, 0, stream>>>(TMP, ppos, fpos, F, F16, total);
  }

  for (int l = 0; l < NLAY; ++l) {
    gemm_qkv<<<dim3(3 * DTOK / 64, NTOK / 32), 32, 0, stream>>>(
        F16, Wqkv16 + (size_t)l * 3 * DTOK * DTOK, bqkv + l * 3 * DTOK, Q16, K16, VT16);
    attn_kernel<<<dim3(SEQ / 16, NHEAD, BS), 32, 0, stream>>>(Q16, K16, VT16, O16);
    gemm_bias_f32<<<dim3(DTOK / 64, NTOK / 32), 32, 0, stream>>>(
        O16, Wo16 + (size_t)l * DTOK * DTOK, bo + l * DTOK, TMP, DTOK, DTOK);
    ln_res_kernel<<<NTOK, 32, 0, stream>>>(F, TMP, ln1s + l * DTOK, ln1b + l * DTOK, F16);
    gemm_bias_gelu_f16<<<dim3(DFF / 64, NTOK / 32), 32, 0, stream>>>(
        F16, W116 + (size_t)l * DFF * DTOK, b1 + l * DFF, H116, DFF, DTOK);
    gemm_bias_f32<<<dim3(DTOK / 64, NTOK / 32), 32, 0, stream>>>(
        H116, W216 + (size_t)l * DTOK * DFF, b2 + l * DTOK, TMP, DTOK, DFF);
    ln_res_kernel<<<NTOK, 32, 0, stream>>>(F, TMP, ln2s + l * DTOK, ln2b + l * DTOK, F16);
  }

  {
    int total = BS * PAIRS * 64 * DTOK;   // 2560 x 512
    pack_init_kernel<<<(total + 255) / 256, 256, 0, stream>>>(F16, APOST, total);
  }
  // head: TMP = APOST @ post_W^T + post_b  (M=2560, N=512, K=512) — reuse TMP
  gemm_bias_f32<<<dim3(DTOK / 64, (BS * PAIRS * 64) / 32), 32, 0, stream>>>(
      APOST, postW16, post_b, TMP, PATCH, DTOK);
  {
    int total = BS * PAIRS * CCH * HW * HW;   // 1,310,720
    depatch_kernel<<<(total + 255) / 256, 256, 0, stream>>>(TMP, out, total);
  }
}
